// model_90812788507092
// MI455X (gfx1250) — compile-verified, matrix operand order checked by reading, not run on hardware
//
#include <hip/hip_runtime.h>
#include <hip/hip_bf16.h>
#include <math.h>

typedef __attribute__((ext_vector_type(16))) _Float16 v16h;
typedef __attribute__((ext_vector_type(8)))  float    v8f;

#define N_NODES 100000
#define N_EDGES 1600000
#define DN 64
#define DE 32
#define HID_ 64

__device__ __forceinline__ v8f wmma32(v16h a, v16h b, v8f c) {
  // D = A(16x32 f16) * B(32x16 f16) + C(16x16 f32)
  return __builtin_amdgcn_wmma_f32_16x16x32_f16(false, a, false, b, (short)0, c, false, false);
}

// Load a 16x32 fragment (A layout; identical lane pattern serves B = W^T when
// src is the row-major [O,K] weight and row0 = output-channel tile base).
// Lane L: r = L&15, half = L>>4; elems 0..7 = src[row0+r][k0+half*8 .. +7],
// elems 8..15 = src[row0+r][k0+16+half*8 .. +7]   (per ISA 16-bit A layout)
__device__ __forceinline__ v16h load_frag(const float* base, int ld, int row0, int k0, int lane) {
  int r = lane & 15, h = lane >> 4;
  const float* p = base + (size_t)(row0 + r) * ld + k0 + h * 8;
  v16h a;
#pragma unroll
  for (int j = 0; j < 8; ++j) a[j] = (_Float16)p[j];
#pragma unroll
  for (int j = 0; j < 8; ++j) a[8 + j] = (_Float16)p[16 + j];
  return a;
}

// LayerNorm over C channels for a 16-row LDS tile; one row per lane (<16).
__device__ __forceinline__ void ln16(float* tile, int C, const float* g, const float* b, int lane) {
  if (lane < 16) {
    float* row = tile + lane * C;
    float m = 0.f;
    for (int c = 0; c < C; ++c) m += row[c];
    m /= (float)C;
    float v = 0.f;
    for (int c = 0; c < C; ++c) { float d = row[c] - m; v += d * d; }
    v /= (float)C;
    float inv = rsqrtf(v + 1e-5f);
    for (int c = 0; c < C; ++c) row[c] = (row[c] - m) * inv * g[c] + b[c];
  }
}

// Order-preserving float<->u32 map so atomicMax(u32) == float max (any sign).
__device__ __forceinline__ unsigned fenc(float f) {
  unsigned u = __float_as_uint(f);
  return (u & 0x80000000u) ? ~u : (u | 0x80000000u);
}
__device__ __forceinline__ float fdec(unsigned u) {
  return (u & 0x80000000u) ? __uint_as_float(u & 0x7FFFFFFFu) : __uint_as_float(~u);
}

// ---- K0: init segment buffers.
__global__ void init_kernel(unsigned* __restrict__ mU, float* __restrict__ sS,
                            float* __restrict__ agg) {
  int i = blockIdx.x * blockDim.x + threadIdx.x;
  agg[i] = 0.f;                       // grid covers exactly N*64
  if (i < N_NODES * 4) { mU[i] = 0u; sS[i] = 0.f; }
}

// ---- K1: Q/K/V/G = x @ W^T (+ bias for G), stored f16. 4 waves/block.
__global__ void node_proj_kernel(const float* __restrict__ x,
                                 const float* __restrict__ WQ, const float* __restrict__ WK,
                                 const float* __restrict__ WV, const float* __restrict__ WG,
                                 const float* __restrict__ ngb,
                                 _Float16* __restrict__ Q, _Float16* __restrict__ K,
                                 _Float16* __restrict__ V, _Float16* __restrict__ G) {
  int wave = threadIdx.x >> 5, lane = threadIdx.x & 31;
  int row0 = blockIdx.x * 16;
  const float* W = (wave == 0) ? WQ : (wave == 1) ? WK : (wave == 2) ? WV : WG;
  _Float16* O = (wave == 0) ? Q : (wave == 1) ? K : (wave == 2) ? V : G;
  v16h a0 = load_frag(x, DN, row0, 0, lane);
  v16h a1 = load_frag(x, DN, row0, 32, lane);
  int n = lane & 15, h = lane >> 4;
#pragma unroll
  for (int t = 0; t < 4; ++t) {
    v16h b0 = load_frag(W, DN, t * 16, 0, lane);
    v16h b1 = load_frag(W, DN, t * 16, 32, lane);
    v8f c = {};
    c = wmma32(a0, b0, c);
    c = wmma32(a1, b1, c);
    int ch = t * 16 + n;
    float bias = ngb[ch];
    if (wave != 3) bias = 0.f;
#pragma unroll
    for (int r = 0; r < 8; ++r)
      O[(size_t)(row0 + r + 8 * h) * HID_ + ch] = (_Float16)(c[r] + bias);
  }
}

// ---- K2: Ef / Eg = edge_attr @ W^T + b (f16 out), fused with per-(edge,head)
//          attention logits + atomic segment-max over dst. 2 waves/block.
__global__ void edge_proj_logits_kernel(const float* __restrict__ ea,
                                        const float* __restrict__ WE, const float* __restrict__ WEb,
                                        const float* __restrict__ WG, const float* __restrict__ WGb,
                                        const _Float16* __restrict__ Q, const _Float16* __restrict__ K,
                                        const int* __restrict__ src, const int* __restrict__ dst,
                                        _Float16* __restrict__ Ef, _Float16* __restrict__ Eg,
                                        float* __restrict__ logits, unsigned* __restrict__ mU) {
  __shared__ _Float16 EfT[16 * 64];
  int wave = threadIdx.x >> 5, lane = threadIdx.x & 31;
  int row0 = blockIdx.x * 16;
  const float* W = wave ? WG : WE;
  const float* B = wave ? WGb : WEb;
  _Float16* O = wave ? Eg : Ef;
  v16h a = load_frag(ea, DE, row0, 0, lane);
  int n = lane & 15, h = lane >> 4;
#pragma unroll
  for (int t = 0; t < 4; ++t) {
    v16h b = load_frag(W, DE, t * 16, 0, lane);
    v8f c = {};
    c = wmma32(a, b, c);
    int ch = t * 16 + n;
#pragma unroll
    for (int r = 0; r < 8; ++r) {
      int rr = r + 8 * h;
      _Float16 v = (_Float16)(c[r] + B[ch]);
      O[(size_t)(row0 + rr) * HID_ + ch] = v;
      if (wave == 0) EfT[rr * HID_ + ch] = v;   // park Ef tile for logit pass
    }
  }
  __syncthreads();
  // 64 threads -> 16 edges x 4 heads: logit = sum_d q*k*ef / sqrt(16)
  int er = threadIdx.x >> 2, hh = threadIdx.x & 3;
  int e = row0 + er;
  int sn = src[e], dn = dst[e];
  const _Float16* q = Q + (size_t)dn * 64 + hh * 16;
  const _Float16* k = K + (size_t)sn * 64 + hh * 16;
  const _Float16* f = EfT + er * 64 + hh * 16;
  float acc = 0.f;
#pragma unroll
  for (int j = 0; j < 16; ++j) acc += (float)q[j] * (float)k[j] * (float)f[j];
  acc *= 0.25f;
  logits[(size_t)e * 4 + hh] = acc;
  atomicMax(&mU[dn * 4 + hh], fenc(acc));
}

// ---- K3: ex = exp(logit - max); atomic segment sum.
__global__ void softmax_kernel(const int* __restrict__ dst, const unsigned* __restrict__ mU,
                               float* __restrict__ logits, float* __restrict__ sS) {
  int i = blockIdx.x * blockDim.x + threadIdx.x;
  int e = i >> 2, hh = i & 3;
  int dn = dst[e];
  float ex = expf(logits[i] - fdec(mU[dn * 4 + hh]));
  logits[i] = ex;
  atomicAdd(&sS[dn * 4 + hh], ex);
}

// ---- K4: gated message + scatter-add aggregation (per edge-channel).
__global__ void message_kernel(const _Float16* __restrict__ G, const _Float16* __restrict__ Eg,
                               const _Float16* __restrict__ V, const float* __restrict__ ex,
                               const float* __restrict__ sS,
                               const int* __restrict__ src, const int* __restrict__ dst,
                               float* __restrict__ agg) {
  size_t i = (size_t)blockIdx.x * blockDim.x + threadIdx.x;   // e*64 + c
  int e = (int)(i >> 6), c = (int)(i & 63), hh = c >> 4;
  int sn = src[e], dn = dst[e];
  float alpha = ex[(size_t)e * 4 + hh] / (sS[dn * 4 + hh] + 1e-16f);
  float gate = 1.f / (1.f + expf(-((float)G[(size_t)sn * 64 + c] + (float)Eg[i])));
  atomicAdd(&agg[(size_t)dn * 64 + c], alpha * gate * (float)V[(size_t)sn * 64 + c]);
}

// ---- K5: node update: agg@WO^T + b + x, LN, FFN, LN. One wave per 16 rows.
__global__ void node_update_kernel(const float* __restrict__ agg, const float* __restrict__ x,
                                   const float* __restrict__ WO, const float* __restrict__ WOb,
                                   const float* __restrict__ w1, const float* __restrict__ b1,
                                   const float* __restrict__ w2, const float* __restrict__ b2,
                                   const float* __restrict__ n1g, const float* __restrict__ n1b,
                                   const float* __restrict__ n2g, const float* __restrict__ n2b,
                                   float* __restrict__ out) {
  __shared__ float U[16 * 64];
  __shared__ float H[16 * 64];
  int lane = threadIdx.x, n = lane & 15, h = lane >> 4;
  int row0 = blockIdx.x * 16;
  v16h a0 = load_frag(agg, 64, row0, 0, lane);
  v16h a1 = load_frag(agg, 64, row0, 32, lane);
#pragma unroll
  for (int t = 0; t < 4; ++t) {
    v16h b0 = load_frag(WO, 64, t * 16, 0, lane);
    v16h bv = load_frag(WO, 64, t * 16, 32, lane);
    v8f c = {}; c = wmma32(a0, b0, c); c = wmma32(a1, bv, c);
    int ch = t * 16 + n;
#pragma unroll
    for (int r = 0; r < 8; ++r) {
      int rr = r + 8 * h;
      U[rr * 64 + ch] = c[r] + WOb[ch] + x[(size_t)(row0 + rr) * 64 + ch];
    }
  }
  __syncthreads();
  ln16(U, 64, n1g, n1b, lane);
  __syncthreads();
  a0 = load_frag(U, 64, 0, 0, lane);
  a1 = load_frag(U, 64, 0, 32, lane);
#pragma unroll
  for (int t = 0; t < 4; ++t) {
    v16h b0 = load_frag(w1, 64, t * 16, 0, lane);
    v16h bv = load_frag(w1, 64, t * 16, 32, lane);
    v8f c = {}; c = wmma32(a0, b0, c); c = wmma32(a1, bv, c);
    int ch = t * 16 + n;
#pragma unroll
    for (int r = 0; r < 8; ++r) {
      int rr = r + 8 * h;
      H[rr * 64 + ch] = fmaxf(c[r] + b1[ch], 0.f);
    }
  }
  __syncthreads();
  a0 = load_frag(H, 64, 0, 0, lane);
  a1 = load_frag(H, 64, 0, 32, lane);
#pragma unroll
  for (int t = 0; t < 4; ++t) {
    v16h b0 = load_frag(w2, 64, t * 16, 0, lane);
    v16h bv = load_frag(w2, 64, t * 16, 32, lane);
    v8f c = {}; c = wmma32(a0, b0, c); c = wmma32(a1, bv, c);
    int ch = t * 16 + n;
#pragma unroll
    for (int r = 0; r < 8; ++r) {
      int rr = r + 8 * h;
      H[rr * 64 + ch] = c[r] + b2[ch] + U[rr * 64 + ch];
    }
  }
  __syncthreads();
  ln16(H, 64, n2g, n2b, lane);
  __syncthreads();
  for (int i = lane; i < 16 * 64; i += 32)
    out[(size_t)row0 * 64 + i] = H[i];
}

// ---- K6: edge update; eij = Q[dst]*K[src]*Ef*0.25 rebuilt on the fly (f16 in, f32 math).
__global__ void edge_update_kernel(const _Float16* __restrict__ Q, const _Float16* __restrict__ K,
                                   const _Float16* __restrict__ Ef, const float* __restrict__ ea,
                                   const int* __restrict__ src, const int* __restrict__ dst,
                                   const float* __restrict__ WOe, const float* __restrict__ WOeb,
                                   const float* __restrict__ w1, const float* __restrict__ b1,
                                   const float* __restrict__ w2, const float* __restrict__ b2,
                                   const float* __restrict__ n1g, const float* __restrict__ n1b,
                                   const float* __restrict__ n2g, const float* __restrict__ n2b,
                                   float* __restrict__ eout) {
  __shared__ float E0[16 * 32];
  __shared__ float H[16 * 64];
  int lane = threadIdx.x, n = lane & 15, h = lane >> 4;
  int row0 = blockIdx.x * 16;
  int e = row0 + (lane & 15);
  int sn = src[e], dn = dst[e];
  const _Float16* q = Q + (size_t)dn * 64;
  const _Float16* kk = K + (size_t)sn * 64;
  const _Float16* f = Ef + (size_t)e * 64;
  v16h a0, a1;
#pragma unroll
  for (int j = 0; j < 8; ++j) {
    int k0 = h * 8 + j;
    a0[j]     = (_Float16)((float)q[k0]      * (float)kk[k0]      * (float)f[k0]      * 0.25f);
    a0[8 + j] = (_Float16)((float)q[k0 + 16] * (float)kk[k0 + 16] * (float)f[k0 + 16] * 0.25f);
    a1[j]     = (_Float16)((float)q[k0 + 32] * (float)kk[k0 + 32] * (float)f[k0 + 32] * 0.25f);
    a1[8 + j] = (_Float16)((float)q[k0 + 48] * (float)kk[k0 + 48] * (float)f[k0 + 48] * 0.25f);
  }
#pragma unroll
  for (int t = 0; t < 2; ++t) {
    v16h b0 = load_frag(WOe, 64, t * 16, 0, lane);
    v16h bv = load_frag(WOe, 64, t * 16, 32, lane);
    v8f c = {}; c = wmma32(a0, b0, c); c = wmma32(a1, bv, c);
    int ch = t * 16 + n;
#pragma unroll
    for (int r = 0; r < 8; ++r) {
      int rr = r + 8 * h;
      E0[rr * 32 + ch] = c[r] + WOeb[ch] + ea[(size_t)(row0 + rr) * 32 + ch];
    }
  }
  __syncthreads();
  ln16(E0, 32, n1g, n1b, lane);
  __syncthreads();
  v16h ae = load_frag(E0, 32, 0, 0, lane);
#pragma unroll
  for (int t = 0; t < 4; ++t) {
    v16h b = load_frag(w1, 32, t * 16, 0, lane);
    v8f c = {}; c = wmma32(ae, b, c);
    int ch = t * 16 + n;
#pragma unroll
    for (int r = 0; r < 8; ++r) {
      int rr = r + 8 * h;
      H[rr * 64 + ch] = fmaxf(c[r] + b1[ch], 0.f);
    }
  }
  __syncthreads();
  a0 = load_frag(H, 64, 0, 0, lane);
  a1 = load_frag(H, 64, 0, 32, lane);
#pragma unroll
  for (int t = 0; t < 2; ++t) {
    v16h b0 = load_frag(w2, 64, t * 16, 0, lane);
    v16h bv = load_frag(w2, 64, t * 16, 32, lane);
    v8f c = {}; c = wmma32(a0, b0, c); c = wmma32(a1, bv, c);
    int ch = t * 16 + n;
#pragma unroll
    for (int r = 0; r < 8; ++r) {
      int rr = r + 8 * h;
      E0[rr * 32 + ch] = c[r] + b2[ch] + E0[rr * 32 + ch];
    }
  }
  __syncthreads();
  ln16(E0, 32, n2g, n2b, lane);
  __syncthreads();
  for (int i = lane; i < 16 * 32; i += 32)
    eout[(size_t)row0 * 32 + i] = E0[i];
}

extern "C" void kernel_launch(void* const* d_in, const int* in_sizes, int n_in,
                              void* d_out, int out_size, void* d_ws, size_t ws_size,
                              hipStream_t stream) {
  const float* x    = (const float*)d_in[0];
  const float* ea   = (const float*)d_in[1];
  const float* WQ   = (const float*)d_in[2];
  const float* WK   = (const float*)d_in[3];
  const float* WV   = (const float*)d_in[4];
  const float* WO   = (const float*)d_in[5];
  const float* WOb  = (const float*)d_in[6];
  const float* WE   = (const float*)d_in[7];
  const float* WEb  = (const float*)d_in[8];
  const float* WOe  = (const float*)d_in[9];
  const float* WOeb = (const float*)d_in[10];
  const float* ngw  = (const float*)d_in[11];
  const float* ngb  = (const float*)d_in[12];
  const float* egw  = (const float*)d_in[13];
  const float* egb  = (const float*)d_in[14];
  const float* fw1  = (const float*)d_in[15];
  const float* fb1  = (const float*)d_in[16];
  const float* fw2  = (const float*)d_in[17];
  const float* fb2  = (const float*)d_in[18];
  const float* few1 = (const float*)d_in[19];
  const float* feb1 = (const float*)d_in[20];
  const float* few2 = (const float*)d_in[21];
  const float* feb2 = (const float*)d_in[22];
  const float* n1g  = (const float*)d_in[23];
  const float* n1b  = (const float*)d_in[24];
  const float* n2g  = (const float*)d_in[25];
  const float* n2b  = (const float*)d_in[26];
  const float* n1eg = (const float*)d_in[27];
  const float* n1eb = (const float*)d_in[28];
  const float* n2eg = (const float*)d_in[29];
  const float* n2eb = (const float*)d_in[30];
  const int*   ei   = (const int*)d_in[31];
  const int* srcI = ei;
  const int* dstI = ei + N_EDGES;

  // Workspace carve-up (f16 intermediates; all counts multiples of 16 -> aligned)
  char* wsb = (char*)d_ws;
  size_t o = 0;
  _Float16* Qh = (_Float16*)(wsb + o); o += (size_t)N_NODES * 64 * 2;
  _Float16* Kh = (_Float16*)(wsb + o); o += (size_t)N_NODES * 64 * 2;
  _Float16* Vh = (_Float16*)(wsb + o); o += (size_t)N_NODES * 64 * 2;
  _Float16* Gh = (_Float16*)(wsb + o); o += (size_t)N_NODES * 64 * 2;
  _Float16* Efh = (_Float16*)(wsb + o); o += (size_t)N_EDGES * 64 * 2;
  _Float16* Egh = (_Float16*)(wsb + o); o += (size_t)N_EDGES * 64 * 2;
  float* lg  = (float*)(wsb + o);    o += (size_t)N_EDGES * 4 * 4;
  unsigned* mU = (unsigned*)(wsb + o); o += (size_t)N_NODES * 4 * 4;
  float* sS  = (float*)(wsb + o);    o += (size_t)N_NODES * 4 * 4;
  float* agg = (float*)(wsb + o);    o += (size_t)N_NODES * 64 * 4;

  float* outN = (float*)d_out;
  float* outE = outN + (size_t)N_NODES * 64;

  init_kernel<<<(N_NODES * 64) / 256, 256, 0, stream>>>(mU, sS, agg);
  node_proj_kernel<<<N_NODES / 16, 128, 0, stream>>>(x, WQ, WK, WV, ngw, ngb, Qh, Kh, Vh, Gh);
  edge_proj_logits_kernel<<<N_EDGES / 16, 64, 0, stream>>>(ea, WE, WEb, egw, egb,
                                                           Qh, Kh, srcI, dstI,
                                                           Efh, Egh, lg, mU);
  softmax_kernel<<<(N_EDGES * 4) / 256, 256, 0, stream>>>(dstI, mU, lg, sS);
  message_kernel<<<(int)(((size_t)N_EDGES * 64) / 256), 256, 0, stream>>>(Gh, Egh, Vh, lg, sS,
                                                                          srcI, dstI, agg);
  node_update_kernel<<<N_NODES / 16, 32, 0, stream>>>(agg, x, WO, WOb, fw1, fb1, fw2, fb2,
                                                      n1g, n1b, n2g, n2b, outN);
  edge_update_kernel<<<N_EDGES / 16, 32, 0, stream>>>(Qh, Kh, Efh, ea, srcI, dstI, WOe, WOeb,
                                                      few1, feb1, few2, feb2,
                                                      n1eg, n1eb, n2eg, n2eb, outE);
}
